// SelfAttention_70025146794123
// MI455X (gfx1250) — matrix operand order repaired and checked by reading, not compiled, not run on hardware
//
#include <hip/hip_runtime.h>
#include <hip/hip_bf16.h>

// ---------------------------------------------------------------------------
// Self-attention (EMB=512, HEADS=8, head_dim=512(!), B=4, T=1024), bf16 WMMA.
//   Q/K/V = x @ W{q,k,v}            : [4096,512] @ [512,4096]  (bf16 wmma)
//   flash attention (causal)        : per (b,h), online softmax
//   out   = O @ Wu + bu             : [4096,4096] @ [4096,512] (bf16 wmma, f32 out)
// Workspace usage: ~148 MiB (bf16 copies of operands + Q/K/V/O).
// CDNA5 paths: v_wmma_f32_16x16x32_bf16, global_load_async_to_lds_b128 (+
// s_wait_asynccnt) for contiguous tile staging, global_prefetch_b8.
// ---------------------------------------------------------------------------

typedef __attribute__((ext_vector_type(16))) __bf16          v16bf;
typedef __attribute__((ext_vector_type(8)))  float           v8f;
typedef __attribute__((ext_vector_type(8)))  unsigned short  ushort8;

// Async-LDS builtin operand types: pointer to 16-byte int vector in the
// global (AS1) / LDS (AS3) address spaces (signature revealed by clang).
typedef int v4i __attribute__((vector_size(16)));
typedef __attribute__((address_space(1))) v4i as1_v4i;
typedef __attribute__((address_space(3))) v4i as3_v4i;

#if __has_builtin(__builtin_amdgcn_global_load_async_to_lds_b128)
#define HAVE_ASYNC_LDS 1
#else
#define HAVE_ASYNC_LDS 0
#endif

union BFV {              // one WMMA 16-bit A/B operand (8 VGPRs)
    v16bf          v;
    ushort8        h[2];
    unsigned short u[16];
};
union F8 {               // one WMMA f32 C/D operand (8 VGPRs)
    v8f   v;
    float f[8];
};

__device__ __forceinline__ unsigned short f32_to_bf16(float f) {
    unsigned int u = __float_as_uint(f);
    unsigned int r = u + 0x7FFFu + ((u >> 16) & 1u);      // round-nearest-even
    if ((u & 0x7F800000u) == 0x7F800000u) r = u;          // inf / NaN: truncate
    return (unsigned short)(r >> 16);
}

__device__ __forceinline__ v8f wmma_bf16(const BFV& a, const BFV& b, v8f c) {
    return __builtin_amdgcn_wmma_f32_16x16x32_bf16(
        /*neg_a=*/false, a.v, /*neg_b=*/false, b.v,
        /*c_mod=*/(short)0, c, /*reuse_a=*/false, /*reuse_b=*/false);
}

// 16-byte global -> LDS copy: async DMA path on CDNA5, sync fallback otherwise.
__device__ __forceinline__ void lds_copy16(unsigned short* ldst,
                                           const unsigned short* gsrc) {
#if HAVE_ASYNC_LDS
    __builtin_amdgcn_global_load_async_to_lds_b128(
        (as1_v4i*)(size_t)gsrc, (as3_v4i*)(size_t)ldst, 0, 0);
#else
    *reinterpret_cast<ushort8*>(ldst) =
        *reinterpret_cast<const ushort8*>(gsrc);
#endif
}

__device__ __forceinline__ void lds_copy_wait() {
#if HAVE_ASYNC_LDS
#if __has_builtin(__builtin_amdgcn_s_wait_asynccnt)
    __builtin_amdgcn_s_wait_asynccnt(0);
#else
    asm volatile("s_wait_asynccnt 0" ::: "memory");
#endif
#endif
}

// ---------------------------------------------------------------------------
// fp32 -> bf16 conversion (grid-stride)
// ---------------------------------------------------------------------------
__global__ void cvt_bf16_kernel(const float* __restrict__ in,
                                unsigned short* __restrict__ out, int n) {
    int i = blockIdx.x * blockDim.x + threadIdx.x;
    int stride = gridDim.x * blockDim.x;
    for (; i < n; i += stride) out[i] = f32_to_bf16(in[i]);
}

// ---------------------------------------------------------------------------
// bf16 GEMM: C[M,N] = A[M,K] @ B[K,N], f32 accum.
// Tile: BM=64 x BN=128 x BK=32, 256 threads = 8 waves (4 M-waves x 2 N-waves),
// each wave computes 16x64 via 4x v_wmma_f32_16x16x32_bf16 per K-step.
// A tile staged with async global->LDS DMA; B tile transposed in LDS ([n][k])
// so B fragments are 2x ds_load_b128.
// Requires M%64==0, N%128==0, K%32==0 (true for all uses here).
// ---------------------------------------------------------------------------
template <bool F32OUT>
__global__ __launch_bounds__(256) void gemm_bf16_kernel(
    const unsigned short* __restrict__ A,   // [M][K] bf16
    const unsigned short* __restrict__ B,   // [K][N] bf16
    unsigned short* __restrict__ Cbf,       // bf16 out (if !F32OUT)
    float* __restrict__ Cf,                 // f32 out (if F32OUT)
    const float* __restrict__ bias,         // may be null
    int M, int N, int K)
{
    __shared__ __align__(16) unsigned short Alds[64][32];    // 4 KB, row-major
    __shared__ __align__(16) unsigned short BldsT[128][32];  // 8 KB, [n][k]

    const int tid  = threadIdx.x;
    const int lane = tid & 31;
    const int wave = tid >> 5;
    const int wm   = wave & 3;      // M sub-tile 0..3
    const int wn   = wave >> 2;     // N sub-tile 0..1
    const int l15  = lane & 15;
    const int half = lane >> 4;
    const int mbase = blockIdx.y * 64;
    const int nbase = blockIdx.x * 128;

    F8 acc[4];
    #pragma unroll
    for (int t = 0; t < 4; ++t) acc[t].v = {};

    for (int k0 = 0; k0 < K; k0 += 32) {
        // ---- stage A tile 64x32 (one async b128 per thread) --------------
        {
            int row = tid >> 2;
            int col = (tid & 3) << 3;
            lds_copy16(&Alds[row][col],
                       &A[(size_t)(mbase + row) * K + k0 + col]);
        }
        // ---- stage B tile 32x128, transposed into LDS --------------------
        #pragma unroll
        for (int rep = 0; rep < 2; ++rep) {
            int c  = tid * 2 + rep;        // 0..511 chunks of 8 bf16
            int k  = c >> 4;               // 0..31
            int n0 = (c & 15) << 3;        // 0..120
            BFV g;
            g.h[0] = *reinterpret_cast<const ushort8*>(
                &B[(size_t)(k0 + k) * N + nbase + n0]);
            #pragma unroll
            for (int j = 0; j < 8; ++j) BldsT[n0 + j][k] = g.u[j];
        }
        // prefetch next A tile row for this thread (global_prefetch_b8 path)
        if (k0 + 32 < K)
            __builtin_prefetch(&A[(size_t)(mbase + (tid >> 2)) * K + k0 + 32], 0, 1);
        lds_copy_wait();
        __syncthreads();

        // ---- A fragment: lane=M row, K split {kb..kb+7, kb+8..kb+15} ----
        BFV a;
        {
            int m  = wm * 16 + l15;
            int kb = half * 16;
            a.h[0] = *reinterpret_cast<const ushort8*>(&Alds[m][kb]);
            a.h[1] = *reinterpret_cast<const ushort8*>(&Alds[m][kb + 8]);
        }
        // ---- 4 WMMA along N ---------------------------------------------
        #pragma unroll
        for (int nt = 0; nt < 4; ++nt) {
            BFV b;
            int n  = wn * 64 + nt * 16 + l15;
            int kk = half * 16;
            b.h[0] = *reinterpret_cast<const ushort8*>(&BldsT[n][kk]);
            b.h[1] = *reinterpret_cast<const ushort8*>(&BldsT[n][kk + 8]);
            acc[nt].v = wmma_bf16(a, b, acc[nt].v);
        }
        __syncthreads();
    }

    // ---- epilogue: C/D layout lane=N, VGPR r -> M = r (+8 for hi half) ---
    #pragma unroll
    for (int nt = 0; nt < 4; ++nt) {
        int n = nbase + wn * 64 + nt * 16 + l15;
        #pragma unroll
        for (int r = 0; r < 8; ++r) {
            int m = mbase + wm * 16 + r + half * 8;
            float v = acc[nt].f[r];
            if constexpr (F32OUT) {
                if (bias) v += bias[n];
                Cf[(size_t)m * N + n] = v;
            } else {
                Cbf[(size_t)m * N + n] = f32_to_bf16(v);
            }
        }
    }
}

// ---------------------------------------------------------------------------
// Flash attention (causal), per (b,h). 128 threads = 4 waves; each wave owns
// 16 q-rows with a full 16x512 f32 accumulator (32x v8f) in registers.
// Q fragments are register-resident; K tile (32x512) staged via async
// global->LDS DMA; V streamed in 128-wide e-chunks (transposed in LDS).
// Online softmax via shfl_xor reductions over the 16-lane halves of the
// WMMA C/D layout.
// ---------------------------------------------------------------------------
__global__ __launch_bounds__(128, 1) void flash_attn_kernel(
    const unsigned short* __restrict__ Q,   // [4096][4096] row=b*T+t, col=h*512+e
    const unsigned short* __restrict__ Km,
    const unsigned short* __restrict__ Vm,
    unsigned short* __restrict__ O)         // [4096][4096] bf16
{
    constexpr int T = 1024, E = 512, HE = 4096;
    __shared__ __align__(16) unsigned short Klds[32][512];    // 32 KB
    __shared__ __align__(16) unsigned short VTlds[128][32];   //  8 KB  [e][kv]
    __shared__ __align__(16) unsigned short Plds[4][16][32];  //  4 KB  per-wave

    const int tid   = threadIdx.x;
    const int lane  = tid & 31;
    const int wave  = tid >> 5;
    const int l15   = lane & 15;
    const int half  = lane >> 4;
    const int bh    = blockIdx.y;        // 0..31
    const int bb    = bh >> 3;
    const int hh    = bh & 7;
    const int qbase = blockIdx.x * 64;
    const int qrow  = qbase + wave * 16;

    // ---- Q fragments (A layout) for e-chunks of 32, register resident ----
    BFV qa[16];
    {
        int m = qrow + l15;
        size_t rbase = (size_t)(bb * T + m) * HE + (size_t)hh * E;
        int kb = half * 16;
        #pragma unroll
        for (int c = 0; c < 16; ++c) {
            qa[c].h[0] = *reinterpret_cast<const ushort8*>(&Q[rbase + c * 32 + kb]);
            qa[c].h[1] = *reinterpret_cast<const ushort8*>(&Q[rbase + c * 32 + kb + 8]);
        }
    }

    F8 o[32];
    #pragma unroll
    for (int i = 0; i < 32; ++i) o[i].v = {};
    float mrun[8], lrun[8];
    #pragma unroll
    for (int r = 0; r < 8; ++r) { mrun[r] = -__builtin_inff(); lrun[r] = 0.f; }

    const float scale = 0.044194173824159216f;   // 1/sqrt(512)
    const int nkv = blockIdx.x * 2 + 2;          // causal: kv blocks of 32

    for (int kvb = 0; kvb < nkv; ++kvb) {
        const int kv0 = kvb * 32;

        // ---- stage K tile [32][512] via async global->LDS ---------------
        #pragma unroll
        for (int i = 0; i < 16; ++i) {
            int c  = tid + i * 128;      // 0..2047 chunks of 8 bf16
            int kr = c >> 6;             // 0..31
            int e0 = (c & 63) << 3;      // 0..504
            lds_copy16(&Klds[kr][e0],
                       &Km[(size_t)(bb * T + kv0 + kr) * HE + hh * E + e0]);
        }
        lds_copy_wait();
        __syncthreads();

        // ---- scores S = Q K^T : 2 n-tiles x 16 k-chunks of WMMA --------
        F8 s0, s1; s0.v = {}; s1.v = {};
        const int kk0 = half * 16;
        #pragma unroll
        for (int c = 0; c < 16; ++c) {
            BFV b0, b1;
            b0.h[0] = *reinterpret_cast<const ushort8*>(&Klds[l15][c * 32 + kk0]);
            b0.h[1] = *reinterpret_cast<const ushort8*>(&Klds[l15][c * 32 + kk0 + 8]);
            b1.h[0] = *reinterpret_cast<const ushort8*>(&Klds[l15 + 16][c * 32 + kk0]);
            b1.h[1] = *reinterpret_cast<const ushort8*>(&Klds[l15 + 16][c * 32 + kk0 + 8]);
            s0.v = wmma_bf16(qa[c], b0, s0.v);
            s1.v = wmma_bf16(qa[c], b1, s1.v);
        }

        // ---- scale + causal mask + online softmax ----------------------
        float fac[8];
        #pragma unroll
        for (int r = 0; r < 8; ++r) {
            int mr = qrow + r + half * 8;
            float v0 = s0.f[r] * scale;
            float v1 = s1.f[r] * scale;
            if (kv0 + l15      > mr) v0 = -__builtin_inff();
            if (kv0 + 16 + l15 > mr) v1 = -__builtin_inff();
            float mx = fmaxf(v0, v1);
            #pragma unroll
            for (int d = 8; d >= 1; d >>= 1)
                mx = fmaxf(mx, __shfl_xor(mx, d, 32));   // stays in 16-lane half
            float mnew = fmaxf(mrun[r], mx);
            float f    = __expf(mrun[r] - mnew);
            float p0   = __expf(v0 - mnew);
            float p1   = __expf(v1 - mnew);
            float sum  = p0 + p1;
            #pragma unroll
            for (int d = 8; d >= 1; d >>= 1)
                sum += __shfl_xor(sum, d, 32);
            lrun[r] = lrun[r] * f + sum;
            mrun[r] = mnew;
            fac[r]  = f;
            s0.f[r] = p0;
            s1.f[r] = p1;
        }
        // rescale O accumulators
        #pragma unroll
        for (int t = 0; t < 32; ++t)
            #pragma unroll
            for (int r = 0; r < 8; ++r) o[t].f[r] *= fac[r];

        // ---- P: D-layout -> LDS -> A-layout ----------------------------
        #pragma unroll
        for (int r = 0; r < 8; ++r) {
            Plds[wave][r + half * 8][l15]      = f32_to_bf16(s0.f[r]);
            Plds[wave][r + half * 8][l15 + 16] = f32_to_bf16(s1.f[r]);
        }
        __syncthreads();
        BFV pa;
        {
            int kb = half * 16;
            pa.h[0] = *reinterpret_cast<const ushort8*>(&Plds[wave][l15][kb]);
            pa.h[1] = *reinterpret_cast<const ushort8*>(&Plds[wave][l15][kb + 8]);
        }

        // ---- O += P @ V, V streamed in e-chunks of 128 -----------------
        #pragma unroll
        for (int ec = 0; ec < 4; ++ec) {
            int e0 = ec * 128;
            #pragma unroll
            for (int i = 0; i < 4; ++i) {
                int c  = tid + i * 128;     // 0..511 chunks of 8 along e
                int kr = c >> 4;            // kv row 0..31
                int n0 = (c & 15) << 3;     // e offset 0..120
                BFV g;
                g.h[0] = *reinterpret_cast<const ushort8*>(
                    &Vm[(size_t)(bb * T + kv0 + kr) * HE + hh * E + e0 + n0]);
                #pragma unroll
                for (int j = 0; j < 8; ++j) VTlds[n0 + j][kr] = g.u[j];
            }
            __syncthreads();
            #pragma unroll
            for (int nt = 0; nt < 8; ++nt) {
                BFV b;
                int n  = nt * 16 + l15;
                int kk = half * 16;
                b.h[0] = *reinterpret_cast<const ushort8*>(&VTlds[n][kk]);
                b.h[1] = *reinterpret_cast<const ushort8*>(&VTlds[n][kk + 8]);
                int oi = ec * 8 + nt;
                o[oi].v = wmma_bf16(pa, b, o[oi].v);
            }
            __syncthreads();
        }
    }

    // ---- normalize and store O (bf16) -----------------------------------
    #pragma unroll
    for (int t = 0; t < 32; ++t) {
        int e = t * 16 + l15;
        #pragma unroll
        for (int r = 0; r < 8; ++r) {
            int row = qrow + r + half * 8;
            float v = o[t].f[r] / lrun[r];
            O[(size_t)(bb * T + row) * HE + hh * E + e] = f32_to_bf16(v);
        }
    }
}

// ---------------------------------------------------------------------------
// Host-side orchestration
// ---------------------------------------------------------------------------
extern "C" void kernel_launch(void* const* d_in, const int* in_sizes, int n_in,
                              void* d_out, int out_size, void* d_ws, size_t ws_size,
                              hipStream_t stream) {
    constexpr int B = 4, T = 1024, E = 512, H = 8;
    constexpr int M  = B * T;     // 4096
    constexpr int HE = H * E;     // 4096

    const float* x  = (const float*)d_in[0];
    const float* Wq = (const float*)d_in[1];
    const float* Wk = (const float*)d_in[2];
    const float* Wv = (const float*)d_in[3];
    const float* Wu = (const float*)d_in[4];
    const float* bu = (const float*)d_in[5];
    float* out = (float*)d_out;

    // workspace layout (bytes) — total ~148 MiB
    char* ws = (char*)d_ws;
    constexpr size_t MiB = 1024ull * 1024ull;
    unsigned short* xb  = (unsigned short*)(ws + 0 * MiB);    // 4096x512
    unsigned short* Wqb = (unsigned short*)(ws + 4 * MiB);    // 512x4096
    unsigned short* Wkb = (unsigned short*)(ws + 8 * MiB);
    unsigned short* Wvb = (unsigned short*)(ws + 12 * MiB);
    unsigned short* Wub = (unsigned short*)(ws + 16 * MiB);   // 4096x512
    unsigned short* Qb  = (unsigned short*)(ws + 20 * MiB);   // 4096x4096
    unsigned short* Kb  = (unsigned short*)(ws + 52 * MiB);
    unsigned short* Vb  = (unsigned short*)(ws + 84 * MiB);
    unsigned short* Ob  = (unsigned short*)(ws + 116 * MiB);

    // 1) fp32 -> bf16 conversions
    cvt_bf16_kernel<<<2048, 256, 0, stream>>>(x,  xb,  M * E);
    cvt_bf16_kernel<<<2048, 256, 0, stream>>>(Wq, Wqb, E * HE);
    cvt_bf16_kernel<<<2048, 256, 0, stream>>>(Wk, Wkb, E * HE);
    cvt_bf16_kernel<<<2048, 256, 0, stream>>>(Wv, Wvb, E * HE);
    cvt_bf16_kernel<<<2048, 256, 0, stream>>>(Wu, Wub, HE * E);

    // 2) projections: Q/K/V = xb @ W*b   (M=4096, K=512, N=4096)
    {
        dim3 grid(HE / 128, M / 64), block(256);
        gemm_bf16_kernel<false><<<grid, block, 0, stream>>>(xb, Wqb, Qb, nullptr, nullptr, M, HE, E);
        gemm_bf16_kernel<false><<<grid, block, 0, stream>>>(xb, Wkb, Kb, nullptr, nullptr, M, HE, E);
        gemm_bf16_kernel<false><<<grid, block, 0, stream>>>(xb, Wvb, Vb, nullptr, nullptr, M, HE, E);
    }

    // 3) causal flash attention per (b,h)
    {
        dim3 grid(T / 64, B * H), block(128);
        flash_attn_kernel<<<grid, block, 0, stream>>>(Qb, Kb, Vb, Ob);
    }

    // 4) output projection: out = Ob @ Wub + bu  (M=4096, K=4096, N=512)
    {
        dim3 grid(E / 128, M / 64), block(256);
        gemm_bf16_kernel<true><<<grid, block, 0, stream>>>(Ob, Wub, nullptr, out, bu, M, E, HE);
    }
}
